// SingleGAtt_52939766890708
// MI455X (gfx1250) — matrix operand order repaired and checked by reading, not compiled, not run on hardware
//
#include <hip/hip_runtime.h>
#include <stdint.h>

// ---------------------------------------------------------------------------
// Problem constants: B=4, S=2048, SV=50, QF=KF=VF=E=1024, SCALE=32
// ---------------------------------------------------------------------------
static constexpr int Bn  = 4;
static constexpr int Sn  = 2048;
static constexpr int SVn = 50;
static constexpr int En  = 1024;
static constexpr float INV_SCALE = 1.0f / 32.0f;

// ---------------------------------------------------------------------------
// bf16 helpers (bit-level; round-to-nearest-even)
// ---------------------------------------------------------------------------
__device__ __forceinline__ uint16_t f2bf_bits(float f) {
    union { float f; uint32_t u; } v; v.f = f;
    uint32_t u = v.u;
    uint32_t r = (u + 0x7FFFu + ((u >> 16) & 1u)) >> 16;
    return (uint16_t)r;
}
__device__ __forceinline__ uint16_t to_bf16(float x)    { return f2bf_bits(x); }
__device__ __forceinline__ uint16_t to_bf16(uint16_t x) { return x; }

typedef __attribute__((ext_vector_type(16))) __bf16 v16bf;
typedef __attribute__((ext_vector_type(8)))  float  v8f;

union Frag16 { v16bf v; uint32_t u[8]; };
union Pack8  { uint4 v; uint32_t w[4]; uint16_t h[8]; };

// ---------------------------------------------------------------------------
// CDNA5 async global->LDS copy (ASYNCcnt-tracked, bypasses VGPRs)
// ---------------------------------------------------------------------------
__device__ __forceinline__ void async_copy_b128(uint32_t lds_off, const void* gptr) {
    asm volatile("global_load_async_to_lds_b128 %0, %1, off"
                 :
                 : "v"(lds_off), "v"((uint64_t)(size_t)gptr)
                 : "memory");
}
__device__ __forceinline__ void wait_asynccnt0() {
    asm volatile("s_wait_asynccnt 0x0" ::: "memory");
}

// ---------------------------------------------------------------------------
// Elementwise tanh + convert f32 -> bf16
// ---------------------------------------------------------------------------
__global__ __launch_bounds__(256)
void tanh_bf16_kernel(const float* __restrict__ x, uint16_t* __restrict__ y, int n) {
    int i = blockIdx.x * 256 + threadIdx.x;
    if (i < n) y[i] = f2bf_bits(tanhf(x[i]));
}

// ---------------------------------------------------------------------------
// LDS-tiled transpose -> bf16.  in: [R,C] (f32 or bf16), out: [C,R] bf16.
// R, C multiples of 32. Batched via blockIdx.z.
// ---------------------------------------------------------------------------
template<typename TIN>
__global__ __launch_bounds__(256)
void transpose_bf16_kernel(const TIN* __restrict__ in, uint16_t* __restrict__ out,
                           int R, int C, long bsIn, long bsOut) {
    __shared__ uint16_t tile[32][33];
    const long bz = blockIdx.z;
    in  += bz * bsIn;
    out += bz * bsOut;
    const int r0 = blockIdx.y * 32, c0 = blockIdx.x * 32;
    const int x  = threadIdx.x & 31, y0 = threadIdx.x >> 5;   // y0: 0..7
#pragma unroll
    for (int j = 0; j < 4; ++j) {
        int r = y0 * 4 + j;
        tile[r][x] = to_bf16(in[(long)(r0 + r) * C + (c0 + x)]);
    }
    __syncthreads();
#pragma unroll
    for (int j = 0; j < 4; ++j) {
        int c = y0 * 4 + j;
        out[(long)(c0 + c) * R + (r0 + x)] = tile[x][c];
    }
}

// ---------------------------------------------------------------------------
// Tiled WMMA GEMM:  C[m][n] = alpha * sum_k A[m][k] * Bt[n][k]  [+ addC]
//   A: [M,K] bf16 row-major (lda);  Bt: [N,K] bf16 row-major (ldb).
//   OUT_BF16 selects C dtype. ADDC adds f32 matrix. ASYNC uses CDNA5
//   global_load_async_to_lds_b128 staging (requires M%128==0 && N%128==0);
//   otherwise register staging with branch-free clamped guards.
// Block 256 threads (8 waves); tile 128x128x32; wave does 2x4 16x16 WMMAs.
// Requirements: K % 32 == 0.
// ---------------------------------------------------------------------------
template<bool OUT_BF16, bool ADDC, bool ASYNC>
__global__ __launch_bounds__(256)
void gemm_wmma_kernel(const uint16_t* __restrict__ Ag, int lda, long bsA,
                      const uint16_t* __restrict__ Bg, int ldb, long bsB,
                      void* __restrict__ Cg, int ldc, long bsC,
                      const float* __restrict__ addC, int ldadd, long bsAdd,
                      int M, int N, int K, float alpha) {
    constexpr int BM = 128, BN = 128, BK = 32, LS = 40;  // LS*2=80B: 16B-aligned rows
    __shared__ alignas(16) uint16_t As[2][BM * LS];
    __shared__ alignas(16) uint16_t Bs[2][BN * LS];

    const int tid  = threadIdx.x;
    const int lane = tid & 31;
    const int wv   = tid >> 5;
    const int wm   = wv >> 1;          // 0..3
    const int wn   = wv & 1;           // 0..1
    const int mTile = blockIdx.y * BM;
    const int nTile = blockIdx.x * BN;
    const long bz   = blockIdx.z;

    Ag += bz * bsA;
    Bg += bz * bsB;

    // staging geometry (identical for A and Bt): chunk i -> row i/4, col (i%4)*8
    const uint16_t* aPtr[2]; const uint16_t* bPtr[2];
    int sRow[2], sCol[2]; bool aOk[2], bOk[2];
#pragma unroll
    for (int it = 0; it < 2; ++it) {
        int i = it * 256 + tid;
        int row = i >> 2, col = (i & 3) * 8;
        sRow[it] = row; sCol[it] = col;
        int gm = mTile + row, gn = nTile + row;
        aOk[it] = (gm < M); bOk[it] = (gn < N);
        aPtr[it] = Ag + (long)(aOk[it] ? gm : 0) * lda + col;
        bPtr[it] = Bg + (long)(bOk[it] ? gn : 0) * ldb + col;
    }

    v8f acc[2][4];
#pragma unroll
    for (int i = 0; i < 2; ++i)
#pragma unroll
        for (int j = 0; j < 4; ++j)
#pragma unroll
            for (int r = 0; r < 8; ++r) acc[i][j][r] = 0.0f;

    const int halfA = lane >> 4;
    const int mrow  = lane & 15;
    const int nlane = lane & 15;
    const int khalf = lane >> 4;
    const int nT    = K / BK;

    Pack8 pa[2], pb[2];
    auto fetchReg = [&](int it, int k0) {
        pa[it].v = *(const uint4*)(aPtr[it] + k0);
        if (!aOk[it]) { pa[it].w[0] = pa[it].w[1] = pa[it].w[2] = pa[it].w[3] = 0; }
        pb[it].v = *(const uint4*)(bPtr[it] + k0);
        if (!bOk[it]) { pb[it].w[0] = pb[it].w[1] = pb[it].w[2] = pb[it].w[3] = 0; }
    };
    auto storeReg = [&](int buf) {
#pragma unroll
        for (int it = 0; it < 2; ++it) {
            *(uint4*)(&As[buf][sRow[it] * LS + sCol[it]]) = pa[it].v;
            *(uint4*)(&Bs[buf][sRow[it] * LS + sCol[it]]) = pb[it].v;
        }
    };
    auto issueAsync = [&](int buf, int k0) {
#pragma unroll
        for (int it = 0; it < 2; ++it) {
            async_copy_b128((uint32_t)(size_t)&As[buf][sRow[it] * LS + sCol[it]],
                            aPtr[it] + k0);
            async_copy_b128((uint32_t)(size_t)&Bs[buf][sRow[it] * LS + sCol[it]],
                            bPtr[it] + k0);
        }
    };

    // ---- stage tile 0 ----
    if (ASYNC) {
        issueAsync(0, 0);
        wait_asynccnt0();
    } else {
        fetchReg(0, 0); fetchReg(1, 0);
        storeReg(0);
    }
    __syncthreads();

    for (int t = 0; t < nT; ++t) {
        const int cur = t & 1;
        if (ASYNC) {
            if (t + 1 < nT) issueAsync(cur ^ 1, (t + 1) * BK);
        } else {
            if (t + 1 < nT) { fetchReg(0, (t + 1) * BK); fetchReg(1, (t + 1) * BK); }
        }
        // ---- compute current tile from LDS ----
        Frag16 af[2];
#pragma unroll
        for (int i = 0; i < 2; ++i) {
            int mbase = wm * 32 + i * 16 + mrow;
#pragma unroll
            for (int g = 0; g < 8; ++g) {
                int kk = (g < 4 ? 2 * g : 16 + 2 * (g - 4)) + halfA * 8;
                af[i].u[g] = *(const uint32_t*)(&As[cur][mbase * LS + kk]);
            }
        }
        Frag16 bf_[4];
#pragma unroll
        for (int j = 0; j < 4; ++j) {
            int nbase = wn * 64 + j * 16 + nlane;
#pragma unroll
            for (int g = 0; g < 8; ++g) {
                int kk = (g < 4 ? 2 * g : 16 + 2 * (g - 4)) + khalf * 8;
                bf_[j].u[g] = *(const uint32_t*)(&Bs[cur][nbase * LS + kk]);
            }
        }
#pragma unroll
        for (int i = 0; i < 2; ++i)
#pragma unroll
            for (int j = 0; j < 4; ++j)
                acc[i][j] = __builtin_amdgcn_wmma_f32_16x16x32_bf16(
                    false, af[i].v, false, bf_[j].v, (short)0, acc[i][j], false, false);

        if (ASYNC) {
            wait_asynccnt0();              // our tile(t+1) copies have landed
        } else {
            if (t + 1 < nT) storeReg(cur ^ 1);
        }
        __syncthreads();
    }

    // ---- epilogue: C/D layout lane = N(0..15) | Mgroup; VGPR r => M = r+8*grp
    const int mgrp = lane >> 4;
#pragma unroll
    for (int i = 0; i < 2; ++i)
#pragma unroll
        for (int j = 0; j < 4; ++j)
#pragma unroll
            for (int r = 0; r < 8; ++r) {
                int gm = mTile + wm * 32 + i * 16 + mgrp * 8 + r;
                int gn = nTile + wn * 64 + j * 16 + nlane;
                if (gm < M && gn < N) {
                    float v = acc[i][j][r] * alpha;
                    if (ADDC) v += addC[bz * bsAdd + (long)gm * ldadd + gn];
                    if (OUT_BF16)
                        ((uint16_t*)Cg)[bz * bsC + (long)gm * ldc + gn] = f2bf_bits(v);
                    else
                        ((float*)Cg)[bz * bsC + (long)gm * ldc + gn] = v;
                }
            }
}

// ---------------------------------------------------------------------------
// Row softmax over 2048 columns, in place on f32; emits bf16 f and bf16 (1-f).
// ---------------------------------------------------------------------------
__global__ __launch_bounds__(256)
void softmax_kernel(float* __restrict__ f, uint16_t* __restrict__ fb,
                    uint16_t* __restrict__ gb) {
    constexpr int COLS = Sn;           // 2048
    constexpr int PER  = COLS / 256;   // 8
    long row = blockIdx.x;
    float*    p   = f  + row * COLS;
    uint16_t* pf  = fb + row * COLS;
    uint16_t* pg  = gb + row * COLS;
    const int tid  = threadIdx.x;
    const int lane = tid & 31;
    const int wv   = tid >> 5;

    __shared__ float red[8];

    float vals[PER];
    float m = -3.402823466e38f;
#pragma unroll
    for (int i = 0; i < PER; ++i) { vals[i] = p[tid + i * 256]; m = fmaxf(m, vals[i]); }
#pragma unroll
    for (int off = 16; off; off >>= 1) m = fmaxf(m, __shfl_xor(m, off, 32));
    if (lane == 0) red[wv] = m;
    __syncthreads();
    float mAll = red[0];
#pragma unroll
    for (int j = 1; j < 8; ++j) mAll = fmaxf(mAll, red[j]);
    __syncthreads();

    float s = 0.0f;
#pragma unroll
    for (int i = 0; i < PER; ++i) { vals[i] = __expf(vals[i] - mAll); s += vals[i]; }
#pragma unroll
    for (int off = 16; off; off >>= 1) s += __shfl_xor(s, off, 32);
    if (lane == 0) red[wv] = s;
    __syncthreads();
    float sAll = 0.0f;
#pragma unroll
    for (int j = 0; j < 8; ++j) sAll += red[j];
    float inv = 1.0f / sAll;
#pragma unroll
    for (int i = 0; i < PER; ++i) {
        float r = vals[i] * inv;
        p [tid + i * 256] = r;
        pf[tid + i * 256] = f2bf_bits(r);
        pg[tid + i * 256] = f2bf_bits(1.0f - r);
    }
}

// ---------------------------------------------------------------------------
// Small K=50 GEMM:  out[row][e] = sum_j vq[row][j] * W[j][e]   (all f32)
// ---------------------------------------------------------------------------
__global__ __launch_bounds__(256)
void smallmm_kernel(const float* __restrict__ vq, const float* __restrict__ W,
                    float* __restrict__ out) {
    int e    = blockIdx.x * 256 + threadIdx.x;   // 0..E-1
    long row = blockIdx.y;                       // 0..B*S-1
    const float* vr = vq + row * SVn;
    float acc = 0.0f;
#pragma unroll 5
    for (int j = 0; j < SVn; ++j) acc += vr[j] * W[(long)j * En + e];
    out[row * (long)En + e] = acc;
}

// ---------------------------------------------------------------------------
// Host orchestration
// ---------------------------------------------------------------------------
static inline int cdiv(int a, int b) { return (a + b - 1) / b; }

extern "C" void kernel_launch(void* const* d_in, const int* in_sizes, int n_in,
                              void* d_out, int out_size, void* d_ws, size_t ws_size,
                              hipStream_t stream) {
    (void)in_sizes; (void)n_in; (void)out_size; (void)ws_size;

    const float* Q    = (const float*)d_in[0];
    const float* K    = (const float*)d_in[1];
    const float* V    = (const float*)d_in[2];
    const float* W_Q  = (const float*)d_in[3];
    const float* W_K  = (const float*)d_in[4];
    const float* W_v  = (const float*)d_in[5];
    const float* W_vq = (const float*)d_in[6];
    const float* W_vk = (const float*)d_in[7];

    const size_t SE  = (size_t)Bn * Sn * En;   // 8,388,608
    const size_t SS  = (size_t)Bn * Sn * Sn;   // 16,777,216
    const size_t SVE = (size_t)Bn * SVn * En;  // 204,800
    const size_t EE  = (size_t)En * En;        // 1,048,576
    const long   bSE = (long)Sn * En;
    const long   bSS = (long)Sn * Sn;
    const long   bVE = (long)SVn * En;
    const long   bSV = (long)Sn * SVn;

    float* out  = (float*)d_out;
    float* Qout = out;                 // [B,S,E]
    float* Kout = out + SE;            // [B,S,E]
    float* Fout = out + 2 * SE;        // [B,S,S]  (f is the 3rd output)

    char* wsp = (char*)d_ws;
    auto alloc = [&](size_t bytes) -> void* {
        void* p = (void*)wsp;
        wsp += (bytes + 255) & ~(size_t)255;
        return p;
    };
    uint16_t* qt   = (uint16_t*)alloc(SE  * 2);  // tanh(Q) bf16; later reused as q2^T
    uint16_t* kt   = (uint16_t*)alloc(SE  * 2);  // tanh(K) bf16; later reused as k2^T
    uint16_t* vt   = (uint16_t*)alloc(SVE * 2);  // tanh(V) bf16
    uint16_t* wqT  = (uint16_t*)alloc(EE  * 2);  // W_Q^T bf16
    uint16_t* wkT  = (uint16_t*)alloc(EE  * 2);
    uint16_t* wvT  = (uint16_t*)alloc(EE  * 2);
    uint16_t* qb   = (uint16_t*)alloc(SE  * 2);  // q bf16
    uint16_t* kb   = (uint16_t*)alloc(SE  * 2);  // k bf16
    uint16_t* vb   = (uint16_t*)alloc(SVE * 2);  // v bf16
    uint16_t* fb   = (uint16_t*)alloc(SS  * 2);  // f bf16
    uint16_t* gb   = (uint16_t*)alloc(SS  * 2);  // (1-f) bf16
    float*    vqB  = (float*)   alloc((size_t)Bn * Sn * SVn * 4);
    float*    vkB  = (float*)   alloc((size_t)Bn * Sn * SVn * 4);
    uint16_t* q2b  = (uint16_t*)alloc(SE  * 2);  // q2 bf16
    uint16_t* k2b  = (uint16_t*)alloc(SE  * 2);  // k2 bf16
    uint16_t* qbT  = (uint16_t*)alloc(SE  * 2);  // q^T bf16 [B,E,S]
    uint16_t* kbT  = (uint16_t*)alloc(SE  * 2);  // k^T bf16 [B,E,S]
    uint16_t* q2bT = qt;                         // alias (qt dead after projections)
    uint16_t* k2bT = kt;                         // alias

    const dim3 blk(256);

    // 1) tanh activations -> bf16; weights -> transposed bf16
    tanh_bf16_kernel<<<cdiv((int)SE, 256),  blk, 0, stream>>>(Q, qt, (int)SE);
    tanh_bf16_kernel<<<cdiv((int)SE, 256),  blk, 0, stream>>>(K, kt, (int)SE);
    tanh_bf16_kernel<<<cdiv((int)SVE, 256), blk, 0, stream>>>(V, vt, (int)SVE);
    {
        dim3 g(En / 32, En / 32, 1);
        transpose_bf16_kernel<float><<<g, blk, 0, stream>>>(W_Q, wqT, En, En, 0, 0);
        transpose_bf16_kernel<float><<<g, blk, 0, stream>>>(W_K, wkT, En, En, 0, 0);
        transpose_bf16_kernel<float><<<g, blk, 0, stream>>>(W_v, wvT, En, En, 0, 0);
    }

    // 2) projections: q = tanh(Q)@W_Q, k = tanh(K)@W_K (async), v (guarded)
    {
        dim3 g(En / 128, (Bn * Sn) / 128, 1);
        gemm_wmma_kernel<true, false, true><<<g, blk, 0, stream>>>(
            qt, En, 0, wqT, En, 0, qb, En, 0, nullptr, 0, 0,
            Bn * Sn, En, En, 1.0f);
        gemm_wmma_kernel<true, false, true><<<g, blk, 0, stream>>>(
            kt, En, 0, wkT, En, 0, kb, En, 0, nullptr, 0, 0,
            Bn * Sn, En, En, 1.0f);
    }
    {
        dim3 g(En / 128, cdiv(Bn * SVn, 128), 1);   // M = 200
        gemm_wmma_kernel<true, false, false><<<g, blk, 0, stream>>>(
            vt, En, 0, wvT, En, 0, vb, En, 0, nullptr, 0, 0,
            Bn * SVn, En, En, 1.0f);
    }

    // 3) q^T, k^T for later A*B^T-style GEMMs
    {
        dim3 g(En / 32, Sn / 32, Bn);
        transpose_bf16_kernel<uint16_t><<<g, blk, 0, stream>>>(qb, qbT, Sn, En, bSE, bSE);
        transpose_bf16_kernel<uint16_t><<<g, blk, 0, stream>>>(kb, kbT, Sn, En, bSE, bSE);
    }

    // 4) qk = q @ k^T / 32 -> f-region (async); vq / vk (N=50, guarded)
    {
        dim3 g(Sn / 128, Sn / 128, Bn);
        gemm_wmma_kernel<false, false, true><<<g, blk, 0, stream>>>(
            qb, En, bSE, kb, En, bSE, Fout, Sn, bSS, nullptr, 0, 0,
            Sn, Sn, En, INV_SCALE);
    }
    {
        dim3 g(1, Sn / 128, Bn);
        gemm_wmma_kernel<false, false, false><<<g, blk, 0, stream>>>(
            qb, En, bSE, vb, En, bVE, vqB, SVn, bSV, nullptr, 0, 0,
            Sn, SVn, En, INV_SCALE);
        gemm_wmma_kernel<false, false, false><<<g, blk, 0, stream>>>(
            kb, En, bSE, vb, En, bVE, vkB, SVn, bSV, nullptr, 0, 0,
            Sn, SVn, En, INV_SCALE);
    }

    // 5) f = softmax(qk) in place; bf16 f and (1-f)
    softmax_kernel<<<Bn * Sn, blk, 0, stream>>>(Fout, fb, gb);

    // 6) q2pre = vq @ W_vq -> park in Qout; k2pre = vk @ W_vk -> park in Kout
    {
        dim3 g(En / 256, Bn * Sn, 1);
        smallmm_kernel<<<g, blk, 0, stream>>>(vqB, W_vq, Qout);
        smallmm_kernel<<<g, blk, 0, stream>>>(vkB, W_vk, Kout);
    }

    // 7) q2 = f @ q + q2pre ; k2 = (1-f) @ k + k2pre   (bf16, async)
    {
        dim3 g(En / 128, Sn / 128, Bn);
        gemm_wmma_kernel<true, true, true><<<g, blk, 0, stream>>>(
            fb, Sn, bSS, qbT, Sn, bSE, q2b, En, bSE, Qout, En, bSE,
            Sn, En, Sn, 1.0f);
        gemm_wmma_kernel<true, true, true><<<g, blk, 0, stream>>>(
            gb, Sn, bSS, kbT, Sn, bSE, k2b, En, bSE, Kout, En, bSE,
            Sn, En, Sn, 1.0f);
    }

    // 8) q2^T, k2^T (into recycled qt/kt regions)
    {
        dim3 g(En / 32, Sn / 32, Bn);
        transpose_bf16_kernel<uint16_t><<<g, blk, 0, stream>>>(q2b, q2bT, Sn, En, bSE, bSE);
        transpose_bf16_kernel<uint16_t><<<g, blk, 0, stream>>>(k2b, k2bT, Sn, En, bSE, bSE);
    }

    // 9) Q_out = f @ q2 ; K_out = (1-f) @ k2   (f32, async, overwrite pre)
    {
        dim3 g(En / 128, Sn / 128, Bn);
        gemm_wmma_kernel<false, false, true><<<g, blk, 0, stream>>>(
            fb, Sn, bSS, q2bT, Sn, bSE, Qout, En, bSE, nullptr, 0, 0,
            Sn, En, Sn, 1.0f);
        gemm_wmma_kernel<false, false, true><<<g, blk, 0, stream>>>(
            gb, Sn, bSS, k2bT, Sn, bSE, Kout, En, bSE, nullptr, 0, 0,
            Sn, En, Sn, 1.0f);
    }
}
